// BolmoLocalLayer_54004918780344
// MI455X (gfx1250) — matrix operand order repaired and checked by reading, not compiled, not run on hardware
//
#include <hip/hip_runtime.h>
#include <hip/hip_bf16.h>
#include <math.h>

#define Bsz 2
#define Ssz 2048
#define Dm  2048
#define Hn  8
#define QKd 1024
#define DQKd 128
#define DVd 256
#define FFd 8192
#define CAPc 15.0f
#define EPSc 1e-6f

typedef __attribute__((ext_vector_type(16))) __bf16 v16bf;
typedef __attribute__((ext_vector_type(8)))  __bf16 v8bf;
typedef __attribute__((ext_vector_type(8)))  float  v8f;

union V16 { v16bf v; v8bf h[2]; };

__device__ __forceinline__ unsigned short f2bf(float f) {
  union { float f; unsigned u; } x; x.f = f;
  unsigned r = x.u + 0x7FFFu + ((x.u >> 16) & 1u);
  return (unsigned short)(r >> 16);
}
__device__ __forceinline__ float bf2f(unsigned short u) {
  union { unsigned u; float f; } x; x.u = ((unsigned)u) << 16;
  return x.f;
}

__device__ __forceinline__ v8f wmma_bf16(v16bf a, v16bf b, v8f c) {
  return __builtin_amdgcn_wmma_f32_16x16x32_bf16(false, a, false, b, (short)0, c,
                                                 false, false);
}

// ---- CDNA5 async global->LDS copy (ASYNCcnt path) via inline asm -----------
// global_load_async_to_lds_b128 vdst(LDS byte offset), vaddr(64-bit), off
__device__ __forceinline__ void async_cp16(const void* g, void* l) {
  unsigned lo = (unsigned)(unsigned long long)l;
  asm volatile("global_load_async_to_lds_b128 %0, %1, off"
               :: "v"(lo), "v"(g) : "memory");
}
__device__ __forceinline__ void async_wait() {
#if defined(__has_builtin) && __has_builtin(__builtin_amdgcn_s_wait_asynccnt)
  __builtin_amdgcn_s_wait_asynccnt(0);
#else
  asm volatile("s_wait_asynccnt 0x0" ::: "memory");
#endif
}

// ---------------- weight convert + transpose: fp32 [K][N] -> bf16 [N][K] ----
__global__ __launch_bounds__(256) void k_wconv(const float* __restrict__ W,
                                               unsigned short* __restrict__ WT,
                                               int K, int N) {
  __shared__ float t[32][33];
  int nt = blockIdx.x * 32, kt = blockIdx.y * 32;
  int tid = threadIdx.x;
  int lk = tid >> 5, ln = tid & 31;
  for (int p = 0; p < 4; ++p)
    t[lk + p * 8][ln] = W[(size_t)(kt + lk + p * 8) * N + nt + ln];
  __syncthreads();
  for (int p = 0; p < 4; ++p)
    WT[(size_t)(nt + lk + p * 8) * K + kt + ln] = f2bf(t[ln][lk + p * 8]);
}

// ---------------- RMSNorm: fp32 in, bf16 out ----------------
__global__ __launch_bounds__(256) void k_rmsnorm(const float* __restrict__ x,
                                                 const float* __restrict__ w,
                                                 unsigned short* __restrict__ out) {
  int row = blockIdx.x;
  const float* xr = x + (size_t)row * Dm;
  unsigned short* orow = out + (size_t)row * Dm;
  int tid = threadIdx.x;
  float vbuf[8];
  float ss = 0.f;
  for (int j = 0; j < 8; ++j) { vbuf[j] = xr[tid + 256 * j]; ss += vbuf[j] * vbuf[j]; }
  for (int m = 1; m < 32; m <<= 1) ss += __shfl_xor(ss, m, 32);
  __shared__ float red[8];
  if ((tid & 31) == 0) red[tid >> 5] = ss;
  __syncthreads();
  float tot = 0.f;
  for (int j = 0; j < 8; ++j) tot += red[j];
  float rs = rsqrtf(tot / (float)Dm + EPSc);
  for (int j = 0; j < 8; ++j)
    orow[tid + 256 * j] = f2bf(w[tid + 256 * j] * vbuf[j] * rs);
}

// ---------------- i/f gate projections + soft-cap (bf16 activations) --------
__global__ __launch_bounds__(256) void k_gates(const unsigned short* __restrict__ hin,
    const float* __restrict__ wig, const float* __restrict__ big,
    const float* __restrict__ wfg, const float* __restrict__ bfg,
    float* __restrict__ ipre, float* __restrict__ fpre) {
  __shared__ float xrow[Dm];
  int row = blockIdx.x;
  int tid = threadIdx.x;
  for (int j = 0; j < 8; ++j)
    xrow[tid + 256 * j] = bf2f(hin[(size_t)row * Dm + tid + 256 * j]);
  __syncthreads();
  int g = tid >> 5, lane = tid & 31;
  float si = 0.f, sf = 0.f;
  for (int d = lane; d < Dm; d += 32) {
    float xv = xrow[d];
    si += xv * wig[d * Hn + g];
    sf += xv * wfg[d * Hn + g];
  }
  for (int m = 1; m < 32; m <<= 1) { si += __shfl_xor(si, m, 32); sf += __shfl_xor(sf, m, 32); }
  if (lane == 0) {
    int b = row / Ssz, s = row % Ssz;
    size_t o = ((size_t)b * Hn + g) * Ssz + s;
    float iv = si + big[g], fv = sf + bfg[g];
    ipre[o] = CAPc * tanhf(iv / CAPc);
    fpre[o] = CAPc * tanhf(fv / CAPc);
  }
}

// ---------------- per-(b,h) scans: Fcum, a = i - F, M = cummax(a), m = F + M --
__global__ __launch_bounds__(256) void k_scan(const float* __restrict__ ipre,
    const float* __restrict__ fpre, float* __restrict__ a_out,
    float* __restrict__ Mx_out, float* __restrict__ m_out) {
  __shared__ float s0[Ssz], s1[Ssz], Fc[Ssz];
  int bh = blockIdx.x, tid = threadIdx.x;
  const float* fr = fpre + (size_t)bh * Ssz;
  const float* ir = ipre + (size_t)bh * Ssz;
  for (int j = tid; j < Ssz; j += 256) {
    float x = fr[j];
    s0[j] = (x < 0.f) ? (x - log1pf(expf(x))) : (-log1pf(expf(-x)));
  }
  __syncthreads();
  float* src = s0; float* dst = s1;
  for (int off = 1; off < Ssz; off <<= 1) {
    for (int j = tid; j < Ssz; j += 256) dst[j] = src[j] + ((j >= off) ? src[j - off] : 0.f);
    __syncthreads();
    float* t = src; src = dst; dst = t;
  }
  for (int j = tid; j < Ssz; j += 256) Fc[j] = src[j];
  __syncthreads();
  for (int j = tid; j < Ssz; j += 256) s0[j] = ir[j] - Fc[j];
  __syncthreads();
  src = s0; dst = s1;
  for (int off = 1; off < Ssz; off <<= 1) {
    for (int j = tid; j < Ssz; j += 256)
      dst[j] = (j >= off) ? fmaxf(src[j], src[j - off]) : src[j];
    __syncthreads();
    float* t = src; src = dst; dst = t;
  }
  for (int j = tid; j < Ssz; j += 256) {
    float a = ir[j] - Fc[j];
    float M = src[j];
    a_out[(size_t)bh * Ssz + j] = a;
    Mx_out[(size_t)bh * Ssz + j] = M;
    m_out[(size_t)bh * Ssz + j] = Fc[j] + M;
  }
}

// ---------------- WMMA bf16 GEMM: OUT[M,N] = A[M,K] @ BT[N,K]^T --------------
// A, BT bf16 (BT pre-transposed: row n holds K contiguous values).
// EPI 0: store bf16   EPI 1: + ADD(fp32), store fp32   EPI 2: silu(g)*u, bf16
template<int EPI>
__global__ __launch_bounds__(256) void k_gemm(
    const unsigned short* __restrict__ A, const unsigned short* __restrict__ B0,
    const unsigned short* __restrict__ B1, const float* __restrict__ ADD,
    void* __restrict__ OUTp, int M, int N, int K) {
  __shared__ alignas(16) unsigned short As[128][32];
  __shared__ alignas(16) unsigned short Bs[128][32];
  __shared__ alignas(16) unsigned short Bs2[(EPI == 2) ? 128 : 1][32];
  int tid = threadIdx.x;
  int wave = tid >> 5, lane = tid & 31;
  int wm = wave >> 1, wn = wave & 1;
  int n0 = blockIdx.x * 128, m0 = blockIdx.y * 128;
  v8f zero = {0.f, 0.f, 0.f, 0.f, 0.f, 0.f, 0.f, 0.f};
  v8f acc[2][4], acc2[2][4];
  for (int i = 0; i < 2; ++i)
    for (int j = 0; j < 4; ++j) { acc[i][j] = zero; acc2[i][j] = zero; }
  int l15 = lane & 15;
  int abase = (lane < 16) ? 0 : 16;
  int bbase = (lane < 16) ? 0 : 16;
  for (int k0 = 0; k0 < K; k0 += 32) {
#pragma unroll
    for (int j = 0; j < 2; ++j) {        // fixed-trip: 2 x 16B chunks/thread
      int c = tid + j * 256;
      int r = c >> 2, q4 = c & 3;
      async_cp16(&A[(size_t)(m0 + r) * K + k0 + q4 * 8], &As[r][q4 * 8]);
      async_cp16(&B0[(size_t)(n0 + r) * K + k0 + q4 * 8], &Bs[r][q4 * 8]);
      if constexpr (EPI == 2)
        async_cp16(&B1[(size_t)(n0 + r) * K + k0 + q4 * 8], &Bs2[r][q4 * 8]);
    }
    async_wait();
    __syncthreads();
    V16 af[2];
    for (int mi = 0; mi < 2; ++mi) {
      int r = wm * 32 + mi * 16 + l15;
      af[mi].h[0] = *(const v8bf*)&As[r][abase];
      af[mi].h[1] = *(const v8bf*)&As[r][abase + 8];
    }
    for (int ni = 0; ni < 4; ++ni) {
      int c = wn * 64 + ni * 16 + l15;
      v16bf bf = *(const v16bf*)&Bs[c][bbase];
      for (int mi = 0; mi < 2; ++mi) acc[mi][ni] = wmma_bf16(af[mi].v, bf, acc[mi][ni]);
      if constexpr (EPI == 2) {
        v16bf bf2 = *(const v16bf*)&Bs2[c][bbase];
        for (int mi = 0; mi < 2; ++mi) acc2[mi][ni] = wmma_bf16(af[mi].v, bf2, acc2[mi][ni]);
      }
    }
    __syncthreads();
  }
  int rhalf = 8 * (lane >> 4);
  for (int mi = 0; mi < 2; ++mi)
    for (int ni = 0; ni < 4; ++ni) {
      int c = n0 + wn * 64 + ni * 16 + l15;
      for (int vg = 0; vg < 8; ++vg) {
        int r = m0 + wm * 32 + mi * 16 + vg + rhalf;
        float val = acc[mi][ni][vg];
        if constexpr (EPI == 1) {
          val += ADD[(size_t)r * N + c];
          ((float*)OUTp)[(size_t)r * N + c] = val;
        } else if constexpr (EPI == 2) {
          float u = acc2[mi][ni][vg];
          val = (val / (1.f + __expf(-val))) * u;
          ((unsigned short*)OUTp)[(size_t)r * N + c] = f2bf(val);
        } else {
          ((unsigned short*)OUTp)[(size_t)r * N + c] = f2bf(val);
        }
      }
    }
}

// ---------------- mLSTM parallel attention (WMMA, bf16 q/k/v) ----------------
__global__ __launch_bounds__(256) void k_attn(
    const unsigned short* __restrict__ q, const unsigned short* __restrict__ kmat,
    const unsigned short* __restrict__ v, const float* __restrict__ a_g,
    const float* __restrict__ Mx_g, const float* __restrict__ m_g,
    float* __restrict__ hat) {
  __shared__ alignas(16) unsigned short Qs[64][DQKd];   // 16 KB
  __shared__ alignas(16) unsigned short Ks[32][DQKd];   // 8 KB
  __shared__ alignas(16) unsigned short Vt[DVd][32];    // 16 KB ([dv][key])
  __shared__ alignas(16) unsigned short Cs[8][16][32];  // 8 KB per-wave C scratch
  __shared__ float mT[64], MxT[64];
  int qt = blockIdx.x, h = blockIdx.y, b = blockIdx.z;
  int qbase = qt * 64;
  int tid = threadIdx.x, wave = tid >> 5, lane = tid & 31;
  int wm = wave >> 1, wn = wave & 1;
  int l15 = lane & 15;
  int abase = (lane < 16) ? 0 : 16;
  int bbase = (lane < 16) ? 0 : 16;
  size_t bhS = ((size_t)b * Hn + h) * Ssz;
  // async-stage Q tile (64 x 128 bf16, K-contiguous)
#pragma unroll
  for (int j = 0; j < 4; ++j) {
    int c = tid + j * 256;
    int r = c >> 4, f = c & 15;
    async_cp16(&q[((size_t)b * Ssz + qbase + r) * QKd + h * DQKd + f * 8], &Qs[r][f * 8]);
  }
  if (tid < 64) { mT[tid] = m_g[bhS + qbase + tid]; MxT[tid] = Mx_g[bhS + qbase + tid]; }
  async_wait();
  __syncthreads();
  v8f zero = {0.f, 0.f, 0.f, 0.f, 0.f, 0.f, 0.f, 0.f};
  v8f hacc[8]; float rsum[8];
  for (int j = 0; j < 8; ++j) { hacc[j] = zero; rsum[j] = 0.f; }
  V16 qf[4];
  {
    int r = wm * 16 + l15;
    for (int kk = 0; kk < 4; ++kk) {
      qf[kk].h[0] = *(const v8bf*)&Qs[r][kk * 32 + abase];
      qf[kk].h[1] = *(const v8bf*)&Qs[r][kk * 32 + abase + 8];
    }
  }
  const float scale = 0.08838834764831845f;  // 1/sqrt(128)
  int nch = qt * 2 + 2;
  for (int kb = 0; kb < nch; ++kb) {
    int kbase = kb * 32;
#pragma unroll
    for (int j = 0; j < 2; ++j) {            // K tile async (32 x 128)
      int c = tid + j * 256;
      int r = c >> 4, f = c & 15;
      async_cp16(&kmat[((size_t)b * Ssz + kbase + r) * QKd + h * DQKd + f * 8],
                 &Ks[r][f * 8]);
    }
    for (int i = 0; i < 32; ++i) {           // V tile, transposed staging
      int idx = tid + i * 256;
      int r = idx >> 8, dv = idx & 255;
      Vt[dv][r] = v[((size_t)b * Ssz + kbase + r) * Dm + h * DVd + dv];
    }
    async_wait();
    __syncthreads();
    v8f sc0 = zero, sc1 = zero;
    for (int kk = 0; kk < 4; ++kk) {
      v16bf b0 = *(const v16bf*)&Ks[l15][kk * 32 + bbase];
      v16bf b1 = *(const v16bf*)&Ks[16 + l15][kk * 32 + bbase];
      sc0 = wmma_bf16(qf[kk].v, b0, sc0);
      sc1 = wmma_bf16(qf[kk].v, b1, sc1);
    }
    for (int vg = 0; vg < 8; ++vg) {
      int rloc = vg + 8 * (lane >> 4);
      int t = qbase + wm * 16 + rloc;
      float Mxt = MxT[wm * 16 + rloc];
      int sA = kbase + l15, sB = kbase + 16 + l15;
      float c0 = 0.f, c1 = 0.f;
      if (sA <= t) c0 = sc0[vg] * scale * __expf(a_g[bhS + sA] - Mxt);
      if (sB <= t) c1 = sc1[vg] * scale * __expf(a_g[bhS + sB] - Mxt);
      rsum[vg] += c0 + c1;
      Cs[wave][rloc][l15] = f2bf(c0);
      Cs[wave][rloc][16 + l15] = f2bf(c1);
    }
    V16 cf;  // wave-private LDS round-trip (DS ops in-order within wave)
    cf.h[0] = *(const v8bf*)&Cs[wave][l15][abase];
    cf.h[1] = *(const v8bf*)&Cs[wave][l15][abase + 8];
    for (int ni = 0; ni < 8; ++ni) {
      int c = wn * 128 + ni * 16 + l15;
      v16bf bv = *(const v16bf*)&Vt[c][bbase];
      hacc[ni] = wmma_bf16(cf.v, bv, hacc[ni]);
    }
    __syncthreads();
  }
  for (int vg = 0; vg < 8; ++vg) {
    float r = rsum[vg];
    r += __shfl_xor(r, 1, 32);
    r += __shfl_xor(r, 2, 32);
    r += __shfl_xor(r, 4, 32);
    r += __shfl_xor(r, 8, 32);
    rsum[vg] = r;
  }
  for (int vg = 0; vg < 8; ++vg) {
    int rloc = vg + 8 * (lane >> 4);
    int t = qbase + wm * 16 + rloc;
    float n = fmaxf(fabsf(rsum[vg]), __expf(-mT[wm * 16 + rloc]));
    float inv = 1.f / n;
    for (int ni = 0; ni < 8; ++ni) {
      int c = h * DVd + wn * 128 + ni * 16 + l15;
      hat[((size_t)b * Ssz + t) * Dm + c] = hacc[ni][vg] * inv;
    }
  }
}

// ------- per-head RMSNorm * mh_w * sigmoid(o_pre): fp32 in -> bf16 out -------
__global__ __launch_bounds__(256) void k_headnorm(
    const float* __restrict__ hat, const unsigned short* __restrict__ opre,
    const float* __restrict__ mhw, unsigned short* __restrict__ outb) {
  int row = blockIdx.x;
  int tid = threadIdx.x, g = tid >> 5, lane = tid & 31;
  float vbuf[8]; float ss = 0.f;
  size_t base = (size_t)row * Dm + g * DVd;
  for (int j = 0; j < 8; ++j) { float x = hat[base + lane + 32 * j]; vbuf[j] = x; ss += x * x; }
  for (int m = 1; m < 32; m <<= 1) ss += __shfl_xor(ss, m, 32);
  float rs = rsqrtf(ss / (float)DVd + EPSc);
  for (int j = 0; j < 8; ++j) {
    int d = lane + 32 * j;
    float og = bf2f(opre[base + d]);
    float sg = 1.f / (1.f + __expf(-og));
    outb[base + d] = f2bf(vbuf[j] * rs * mhw[g * DVd + d] * sg);
  }
}

extern "C" void kernel_launch(void* const* d_in, const int* in_sizes, int n_in,
                              void* d_out, int out_size, void* d_ws, size_t ws_size,
                              hipStream_t stream) {
  const float* x     = (const float*)d_in[0];
  const float* n1w   = (const float*)d_in[1];
  const float* wq    = (const float*)d_in[2];
  const float* wk    = (const float*)d_in[3];
  const float* wv    = (const float*)d_in[4];
  const float* wig   = (const float*)d_in[5];
  const float* big   = (const float*)d_in[6];
  const float* wfg   = (const float*)d_in[7];
  const float* bfg   = (const float*)d_in[8];
  const float* wog   = (const float*)d_in[9];
  const float* mhw   = (const float*)d_in[10];
  const float* wout  = (const float*)d_in[11];
  const float* n2w   = (const float*)d_in[12];
  const float* wgate = (const float*)d_in[13];
  const float* wup   = (const float*)d_in[14];
  const float* wdown = (const float*)d_in[15];
  float* out = (float*)d_out;

  char* wsb = (char*)d_ws;
  const size_t MB = 1024u * 1024u;
  typedef unsigned short us;
  us*    h_in_bf = (us*)(wsb + 0);          // 16 MB  (also h2 later)
  us*    qb      = (us*)(wsb + 16 * MB);    // 8 MB
  us*    kbuf    = (us*)(wsb + 24 * MB);    // 8 MB
  us*    vb      = (us*)(wsb + 32 * MB);    // 16 MB
  us*    opre    = (us*)(wsb + 48 * MB);    // 16 MB
  float* hat     = (float*)(wsb + 64 * MB); // 32 MB
  us*    hat_bf  = (us*)(wsb + 96 * MB);    // 16 MB
  float* x1      = (float*)(wsb + 112 * MB);// 32 MB
  us*    mlp     = (us*)(wsb + 16 * MB);    // 64 MB overlay (q..hat dead by then)
  us*    wA      = (us*)(wsb + 144 * MB);   // 32 MB weight scratch
  us*    wB      = (us*)(wsb + 176 * MB);   // 32 MB weight scratch
  float* ipre    = (float*)(wsb + 208 * MB);
  float* fpre    = ipre + (size_t)Bsz * Hn * Ssz;
  float* a_g     = fpre + (size_t)Bsz * Hn * Ssz;
  float* Mx_g    = a_g + (size_t)Bsz * Hn * Ssz;
  float* m_g     = Mx_g + (size_t)Bsz * Hn * Ssz;

  const unsigned MT = Bsz * Ssz;  // 4096 rows
  dim3 blk(256);

  k_rmsnorm<<<MT, blk, 0, stream>>>(x, n1w, h_in_bf);

  k_wconv<<<dim3(32, 64), blk, 0, stream>>>(wq, wA, 2048, 1024);
  k_gemm<0><<<dim3(8, 32), blk, 0, stream>>>(h_in_bf, wA, nullptr, nullptr, qb, 4096, 1024, 2048);
  k_wconv<<<dim3(32, 64), blk, 0, stream>>>(wk, wA, 2048, 1024);
  k_gemm<0><<<dim3(8, 32), blk, 0, stream>>>(h_in_bf, wA, nullptr, nullptr, kbuf, 4096, 1024, 2048);
  k_wconv<<<dim3(64, 64), blk, 0, stream>>>(wv, wA, 2048, 2048);
  k_gemm<0><<<dim3(16, 32), blk, 0, stream>>>(h_in_bf, wA, nullptr, nullptr, vb, 4096, 2048, 2048);
  k_wconv<<<dim3(64, 64), blk, 0, stream>>>(wog, wA, 2048, 2048);
  k_gemm<0><<<dim3(16, 32), blk, 0, stream>>>(h_in_bf, wA, nullptr, nullptr, opre, 4096, 2048, 2048);

  k_gates<<<MT, blk, 0, stream>>>(h_in_bf, wig, big, wfg, bfg, ipre, fpre);
  k_scan<<<Bsz * Hn, blk, 0, stream>>>(ipre, fpre, a_g, Mx_g, m_g);
  k_attn<<<dim3(Ssz / 64, Hn, Bsz), blk, 0, stream>>>(qb, kbuf, vb, a_g, Mx_g, m_g, hat);
  k_headnorm<<<MT, blk, 0, stream>>>(hat, opre, mhw, hat_bf);

  k_wconv<<<dim3(64, 64), blk, 0, stream>>>(wout, wA, 2048, 2048);
  k_gemm<1><<<dim3(16, 32), blk, 0, stream>>>(hat_bf, wA, nullptr, x, x1, 4096, 2048, 2048);

  k_rmsnorm<<<MT, blk, 0, stream>>>(x1, n2w, h_in_bf);
  k_wconv<<<dim3(256, 64), blk, 0, stream>>>(wgate, wA, 2048, 8192);
  k_wconv<<<dim3(256, 64), blk, 0, stream>>>(wup, wB, 2048, 8192);
  k_gemm<2><<<dim3(64, 32), blk, 0, stream>>>(h_in_bf, wA, wB, nullptr, mlp, 4096, 8192, 2048);
  k_wconv<<<dim3(64, 256), blk, 0, stream>>>(wdown, wA, 8192, 2048);
  k_gemm<1><<<dim3(16, 32), blk, 0, stream>>>(mlp, wA, nullptr, x1, out, 4096, 2048, 8192);
}
